// TemporalBiMambaEncoder_52913997086996
// MI455X (gfx1250) — matrix operand order repaired and checked by reading, not compiled, not run on hardware
//
#include <hip/hip_runtime.h>
#include <cstdint>
#include <cstddef>

// ---------------------------------------------------------------------------
// TemporalBiMambaEncoder for MI455X (gfx1250, wave32, WMMA + TDM).
// All GEMMs -> V_WMMA_F32_16X16X32_BF16 (f32 accum), tiles staged to LDS by
// the Tensor Data Mover (double-buffered, TENSORcnt-tracked). Activations
// live in L2 (~110MB scratch < 192MB L2): compute-bound, not HBM-bound.
// ---------------------------------------------------------------------------

#define BB 4
#define CC 256
#define TT 2048
#define MM (BB*TT)          // 8192 tokens
#define DIN 512
#define DSTATE 16
#define DCONV 4
#define DTRANK 16
#define GN_GROUPS 4
#define EPSF 1e-5f

// --- TDM availability / arity detection (ROCm7.2 = 5 args; therock = 6) ---
#if defined(__has_builtin)
#  if __has_builtin(__builtin_amdgcn_tensor_load_to_lds)
#    define HAVE_TDM 1
#  endif
#endif
#ifndef HAVE_TDM
#  define HAVE_TDM 0
#endif
#if __has_include(<hip/amd_detail/amd_gfx1250_TDM.h>)
#  define TDM_ARITY6 1
#else
#  define TDM_ARITY6 0
#endif

typedef __attribute__((ext_vector_type(16))) __bf16 bf16x16;
typedef __attribute__((ext_vector_type(8)))  __bf16 bf16x8;
typedef __attribute__((ext_vector_type(8)))  float  f32x8;
typedef unsigned int v4u __attribute__((ext_vector_type(4)));
typedef int          v8i __attribute__((ext_vector_type(8)));
typedef int          v4i __attribute__((ext_vector_type(4)));

__device__ __forceinline__ float  bf2f(__bf16 v){ return (float)v; }
__device__ __forceinline__ __bf16 f2bf(float v){ return (__bf16)v; }
__device__ __forceinline__ float  siluf(float x){ return x / (1.f + __expf(-x)); }
__device__ __forceinline__ int    imax(int a, int b){ return a > b ? a : b; }

__device__ __forceinline__ bf16x8 bfzero8(){
  bf16x8 z;
#pragma unroll
  for (int i=0;i<8;i++) z[i] = (__bf16)0.0f;
  return z;
}

#if HAVE_TDM
// ---------------------------------------------------------------------------
// Issue one TDM 2D-tile load (bf16 elements) global -> LDS.
//   D# group0: count=1, lds_addr, 57b global_addr, type=2
//   D# group1: data_size=2B; pad_enable, pad_interval=3 (16 DW = one 32-elt
//              row), pad_amount=3 (4 DW = 8 halves)  => LDS rows of 40 halves,
//              matching the conflict-free fragment layout below.
//   dims are "remaining extent from tile start": rows/cols past the tensor
//   edge are zero-filled by the TDM (handles K=16 and N=48 tiles).
// ---------------------------------------------------------------------------
__device__ __forceinline__ void tdm_load_2d_bf16(
    unsigned int lds_off, const void* gptr,
    unsigned int dim0_rem, unsigned int dim1_rem,
    unsigned int tile0, unsigned int tile1, unsigned int stride0)
{
  unsigned long long ga = (unsigned long long)(uintptr_t)gptr;
  v4u g0;
  g0[0] = 1u;                                        // count=1 (valid user D#)
  g0[1] = lds_off;                                   // lds_addr (bytes)
  g0[2] = (unsigned int)(ga & 0xffffffffu);          // global_addr[31:0]
  g0[3] = (unsigned int)((ga >> 32) & 0x1ffffffu)    // global_addr[56:32]
          | 0x80000000u;                             // type=2 ("image")
  v8i g1;
  g1[0] = (int)((1u << 16)        // data_size = 1 -> 2 bytes
              | (1u << 20)        // pad_enable
              | (3u << 22)        // pad_interval: 16 DWORDs
              | (3u << 25));      // pad_amount:   4 DWORDs
  g1[1] = (int)((dim0_rem & 0xffffu) << 16);                          // dim0 lo16
  g1[2] = (int)(((dim0_rem >> 16) & 0xffffu) | ((dim1_rem & 0xffffu) << 16));
  g1[3] = (int)(((dim1_rem >> 16) & 0xffffu) | (tile0 << 16));        // tile_dim0
  g1[4] = (int)(tile1 & 0xffffu);                                     // tile_dim1
  g1[5] = (int)stride0;                                               // stride0 lo32
  g1[6] = 0;                                                          // stride0 hi / stride1 lo
  g1[7] = 0;
  v4i z4 = {0,0,0,0};
#if TDM_ARITY6
  v8i z8 = {0,0,0,0,0,0,0,0};
  __builtin_amdgcn_tensor_load_to_lds(g0, g1, z4, z4, z8, 0);
#else
  __builtin_amdgcn_tensor_load_to_lds(g0, g1, z4, z4, 0);
#endif
}
#endif // HAVE_TDM

// ---------------------------------------------------------------------------
// f32 -> bf16 cast (weights)
// ---------------------------------------------------------------------------
__global__ void castK(const float* __restrict__ s, __bf16* __restrict__ d, int n){
  int i = blockIdx.x*256 + threadIdx.x;
  if (i < n) d[i] = f2bf(s[i]);
}

// ---------------------------------------------------------------------------
// Conv-stack pass 1: grouped conv (groups of 4 ch, k=3, pad 1) + bias,
// plus GroupNorm sum/sumsq accumulation via wave reduce + global f32 atomics.
// grid (T/256, C, B), block 256
// ---------------------------------------------------------------------------
__global__ __launch_bounds__(256) void convgnK(
    const float* __restrict__ src, float* __restrict__ dst,
    const float* __restrict__ w,  const float* __restrict__ bias,
    float* __restrict__ stats)
{
  const int t = blockIdx.x*256 + threadIdx.x;
  const int c = blockIdx.y;
  const int b = blockIdx.z;
  const int g4 = (c >> 2) << 2;
  float acc = bias[c];
#pragma unroll
  for (int ic = 0; ic < 4; ++ic) {
    const float* xr = src + ((size_t)b*CC + g4 + ic)*TT;
    const float* wr = w + (c*4 + ic)*3;
#pragma unroll
    for (int j = 0; j < 3; ++j) {
      int ts = t - 1 + j;
      if (ts >= 0 && ts < TT) acc += wr[j] * xr[ts];
    }
  }
  dst[((size_t)b*CC + c)*TT + t] = acc;
  float s1 = acc, s2 = acc*acc;
#pragma unroll
  for (int o = 16; o > 0; o >>= 1) {
    s1 += __shfl_xor(s1, o, 32);
    s2 += __shfl_xor(s2, o, 32);
  }
  if ((threadIdx.x & 31) == 0) {
    int gg = c >> 6;   // group of 64 channels
    atomicAdd(&stats[(b*GN_GROUPS + gg)*2 + 0], s1);
    atomicAdd(&stats[(b*GN_GROUPS + gg)*2 + 1], s2);
  }
}

// ---------------------------------------------------------------------------
// Conv-stack pass 2: normalize, affine, LeakyReLU(0.2), mask. In place.
// ---------------------------------------------------------------------------
__global__ __launch_bounds__(256) void gnormK(
    float* __restrict__ buf, const float* __restrict__ stats,
    const float* __restrict__ gnw, const float* __restrict__ gnb,
    const unsigned char* __restrict__ mask)
{
  size_t idx = (size_t)blockIdx.x*256 + threadIdx.x;   // B*C*T
  int t = (int)(idx & (TT-1));
  size_t rest = idx >> 11;
  int c = (int)(rest & (CC-1));
  int b = (int)(rest >> 8);
  const float inv = 1.f / (64.f * (float)TT);
  float s  = stats[(b*GN_GROUPS + (c>>6))*2 + 0];
  float q  = stats[(b*GN_GROUPS + (c>>6))*2 + 1];
  float mu = s*inv;
  float var = q*inv - mu*mu;
  float v = buf[idx];
  v = (v - mu) * rsqrtf(var + EPSF) * gnw[c] + gnb[c];
  v = (v >= 0.f) ? v : 0.2f*v;
  if (mask[b*TT + t]) v = 0.f;
  buf[idx] = v;
}

// ---------------------------------------------------------------------------
// (B,C,T) f32 -> (B*T, C) bf16, LDS-tiled transpose. grid (T/32, C/32, B)
// ---------------------------------------------------------------------------
__global__ __launch_bounds__(256) void transposeK(
    const float* __restrict__ src, __bf16* __restrict__ dst)
{
  __shared__ float tile[32][33];
  const int tx = threadIdx.x, ty = threadIdx.y;  // block (32,8)
  const int t0 = blockIdx.x*32, c0 = blockIdx.y*32, b = blockIdx.z;
#pragma unroll
  for (int k = 0; k < 4; ++k)
    tile[ty + k*8][tx] = src[((size_t)b*CC + c0 + ty + k*8)*TT + t0 + tx];
  __syncthreads();
#pragma unroll
  for (int k = 0; k < 4; ++k) {
    int t = t0 + ty + k*8;
    dst[((size_t)b*TT + t)*CC + c0 + tx] = f2bf(tile[tx][ty + k*8]);
  }
}

// ---------------------------------------------------------------------------
// Generic bf16 WMMA GEMM:  out = epilogue( A(MxK) * W(NxK)^T )
// Block 256 thr = 8 waves; block tile 128x64; wave tile 32x32 (2x2 WMMA).
// Tiles staged to LDS by TDM (double-buffered) when available; LDS rows
// padded to 40 halves -> conflict-free b128 ds fragment loads.
// ---------------------------------------------------------------------------
struct GemmP {
  const __bf16* A; int lda;
  const __bf16* W; int ldw;
  float*  outF;          // f32 output (or null)
  __bf16* outB;          // bf16 output (or null)
  int ldo;
  const float* bias;     // per-N bias or null
  const __bf16* mul;     // elementwise multiplier (per m,n) or null
  int ldmul;
  float* outT;           // transposed masked store (final) or null
  const unsigned char* mask;
  int Tdim;
  int M, N, K;
  int act;               // 0 none, 1 softplus, 2 silu
  int accAdd;            // outF += result
};

__global__ __launch_bounds__(256) void gemmK(GemmP p)
{
#if HAVE_TDM
  __shared__ __align__(16) __bf16 lsA[2][128*40];
  __shared__ __align__(16) __bf16 lsB[2][64*40];
#else
  __shared__ __align__(16) __bf16 lsA[1][128*40];
  __shared__ __align__(16) __bf16 lsB[1][64*40];
#endif
  const int tid  = threadIdx.x;
  const int lane = tid & 31, wid = tid >> 5;
  const int mBase = blockIdx.x * 128;
  const int nBase = blockIdx.y * 64;
  const int waveM = (wid >> 1) * 32;   // 4 waves along M
  const int waveN = (wid & 1)  * 32;   // 2 waves along N

  f32x8 acc[2][2];
#pragma unroll
  for (int i=0;i<2;i++)
#pragma unroll
    for (int j=0;j<2;j++)
#pragma unroll
      for (int r=0;r<8;r++) acc[i][j][r] = 0.f;

#if HAVE_TDM
  auto issueTiles = [&](int k0, int buf) {
    // A tile: 128 rows x 32 cols, zero-filled past (M - mBase, K - k0)
    tdm_load_2d_bf16((unsigned int)(uintptr_t)&lsA[buf][0],
                     p.A + (size_t)mBase * p.lda + k0,
                     (unsigned int)imax(p.K - k0, 0),
                     (unsigned int)imax(p.M - mBase, 0),
                     32u, 128u, (unsigned int)p.lda);
    // W tile: 64 rows x 32 cols, zero-filled past (N - nBase, K - k0)
    tdm_load_2d_bf16((unsigned int)(uintptr_t)&lsB[buf][0],
                     p.W + (size_t)nBase * p.ldw + k0,
                     (unsigned int)imax(p.K - k0, 0),
                     (unsigned int)imax(p.N - nBase, 0),
                     32u, 64u, (unsigned int)p.ldw);
  };
  // prologue: wave 0 issues first tiles, waits TENSORcnt, hands off at barrier
  if (tid < 32) {
    issueTiles(0, 0);
    __builtin_amdgcn_s_wait_tensorcnt(0);
  }
  __syncthreads();
  int cur = 0;
#endif

  for (int k0 = 0; k0 < p.K; k0 += 32) {
#if HAVE_TDM
    // kick off next tile DMA while this one is consumed
    if (tid < 32 && (k0 + 32) < p.K) issueTiles(k0 + 32, cur ^ 1);
    const __bf16* tA = &lsA[cur][0];
    const __bf16* tB = &lsB[cur][0];
#else
    // fallback: cooperative guarded staging (round-1 path)
#pragma unroll
    for (int rep = 0; rep < 2; ++rep) {
      int c = tid + rep*256;
      int row = c >> 2, colh = (c & 3) * 8;
      bf16x8 v = bfzero8();
      int gr = mBase + row, gk = k0 + colh;
      if (gr < p.M && gk + 8 <= p.K)
        v = *(const bf16x8*)(p.A + (size_t)gr * p.lda + gk);
      *(bf16x8*)(&lsA[0][0] + row*40 + colh) = v;
    }
    {
      int row = tid >> 2, colh = (tid & 3) * 8;
      bf16x8 v = bfzero8();
      int gr = nBase + row, gk = k0 + colh;
      if (gr < p.N && gk + 8 <= p.K)
        v = *(const bf16x8*)(p.W + (size_t)gr * p.ldw + gk);
      *(bf16x8*)(&lsB[0][0] + row*40 + colh) = v;
    }
    __syncthreads();
    const __bf16* tA = &lsA[0][0];
    const __bf16* tB = &lsB[0][0];
#endif

    // Fragments per ISA layout tables (CDNA5 ISA):
    //  A 16x32:  lane<16 -> row=lane,    K halves {0..7, 16..23}
    //            lane>=16-> row=lane-16, K halves {8..15,24..31}
    //  B 32x16:  lane<16 -> col=lane,    K {0..7, 16..23}; lane>=16 -> K {8..15,24..31}
    bf16x16 afr[2], bfr[2];
    {
      int r  = lane & 15;
      int ka = (lane < 16) ? 0 : 8;
#pragma unroll
      for (int i=0;i<2;i++) {
        const __bf16* base = tA + (waveM + i*16 + r)*40 + ka;
        bf16x8 lo = *(const bf16x8*)(base);
        bf16x8 hi = *(const bf16x8*)(base + 16);
        afr[i] = __builtin_shufflevector(lo, hi,
                   0,1,2,3,4,5,6,7,8,9,10,11,12,13,14,15);
      }
      int kb = (lane < 16) ? 0 : 8;
#pragma unroll
      for (int j=0;j<2;j++) {
        const __bf16* base = tB + (waveN + j*16 + r)*40 + kb;
        bf16x8 lo = *(const bf16x8*)(base);
        bf16x8 hi = *(const bf16x8*)(base + 16);
        bfr[j] = __builtin_shufflevector(lo, hi,
                   0,1,2,3,4,5,6,7,8,9,10,11,12,13,14,15);
      }
    }
#pragma unroll
    for (int i=0;i<2;i++)
#pragma unroll
      for (int j=0;j<2;j++)
        acc[i][j] = __builtin_amdgcn_wmma_f32_16x16x32_bf16(
            false, afr[i], false, bfr[j], (short)0, acc[i][j], false, false);

#if HAVE_TDM
    __syncthreads();                          // all waves done reading cur
    if (tid < 32) __builtin_amdgcn_s_wait_tensorcnt(0);
    __syncthreads();                          // next buffer visible to all
    cur ^= 1;
#else
    __syncthreads();
#endif
  }

  // Epilogue. C/D layout: VGPR r -> row = r (lane<16) / r+8 (lane>=16), col = lane%16
#pragma unroll
  for (int i=0;i<2;i++) {
#pragma unroll
    for (int j=0;j<2;j++) {
      int n = nBase + waveN + j*16 + (lane & 15);
      if (n >= p.N) continue;
#pragma unroll
      for (int r=0;r<8;r++) {
        int mrow = mBase + waveM + i*16 + ((lane < 16) ? r : r + 8);
        if (mrow >= p.M) continue;
        float v = acc[i][j][r];
        if (p.bias) v += p.bias[n];
        if (p.act == 1)      v = (v > 20.f) ? v : log1pf(__expf(v));
        else if (p.act == 2) v = siluf(v);
        if (p.mul)  v *= bf2f(p.mul[(size_t)mrow * p.ldmul + n]);
        if (p.outT) {
          int bb = mrow / p.Tdim, tt = mrow - bb*p.Tdim;
          float o = p.mask[bb*p.Tdim + tt] ? 0.f : v;
          p.outT[((size_t)bb * p.N + n) * p.Tdim + tt] = o;
        } else if (p.outF) {
          size_t idx = (size_t)mrow * p.ldo + n;
          if (p.accAdd) v += p.outF[idx];
          p.outF[idx] = v;
        } else {
          p.outB[(size_t)mrow * p.ldo + n] = f2bf(v);
        }
      }
    }
  }
}

// ---------------------------------------------------------------------------
// Mamba depthwise causal conv (DCONV=4) + SiLU.  rev=1: anti-causal
// (equivalent to causal conv on the time-flipped sequence).
// xz: (B*T, 2*DIN) f32 (first DIN cols = conv input). xi: (B*T, DIN) bf16.
// ---------------------------------------------------------------------------
__global__ __launch_bounds__(256) void mconvK(
    const float* __restrict__ xz, const float* __restrict__ cw,
    const float* __restrict__ cb, __bf16* __restrict__ xi, int rev)
{
  size_t idx = (size_t)blockIdx.x*256 + threadIdx.x;  // M * DIN
  int d = (int)(idx & (DIN-1));
  int mtok = (int)(idx >> 9);
  int t = mtok & (TT-1);
  int b = mtok >> 11;
  float acc = cb[d];
#pragma unroll
  for (int j = 0; j < DCONV; ++j) {
    int ts = rev ? (t + (DCONV-1) - j) : (t - (DCONV-1) + j);
    if (ts >= 0 && ts < TT)
      acc += cw[d*DCONV + j] * xz[((size_t)(b*TT + ts))*(2*DIN) + d];
  }
  xi[idx] = f2bf(siluf(acc));
}

// ---------------------------------------------------------------------------
// Selective scan. One thread per (b, d) channel, h[16] in VGPRs.
// B/C (32 bf16 per timestep, shared across channels) staged in LDS chunks.
// Fused epilogue: yz = (y + u*D) * silu(z)  ->  bf16 (input to out_proj GEMM)
// grid 8 = B * (DIN/256), block 256.
// ---------------------------------------------------------------------------
#define SCHUNK 64
__global__ __launch_bounds__(256) void scanK(
    const __bf16* __restrict__ proj,  // (M,48): [0:16) dt-rank, [16:32) B, [32:48) C
    const float*  __restrict__ dt,    // (M,DIN)
    const __bf16* __restrict__ xi,    // (M,DIN) u
    const float*  __restrict__ xz,    // (M,2*DIN): cols [DIN,2*DIN) = z
    const float*  __restrict__ A_log, // (DIN,16)
    const float*  __restrict__ Dp,    // (DIN)
    __bf16* __restrict__ yz,          // (M,DIN)
    int rev)
{
  __shared__ __align__(16) __bf16 lsBC[SCHUNK*32];
  const int tid = threadIdx.x;
  const int b = blockIdx.x >> 1;
  const int d = ((blockIdx.x & 1) << 8) + tid;

  float Av[DSTATE];
#pragma unroll
  for (int n = 0; n < DSTATE; ++n) Av[n] = -__expf(A_log[d*DSTATE + n]);
  const float Dd = Dp[d];

  float h[DSTATE];
#pragma unroll
  for (int n = 0; n < DSTATE; ++n) h[n] = 0.f;

  for (int s0 = 0; s0 < TT; s0 += SCHUNK) {
    // cooperative stage of B/C for this chunk (2048 halves, 8 per thread)
    {
      int tl = tid >> 2, colh = (tid & 3) * 8;
      int sp = rev ? (TT - 1 - (s0 + tl)) : (s0 + tl);
      bf16x8 v = *(const bf16x8*)(proj + ((size_t)(b*TT + sp))*48 + 16 + colh);
      *(bf16x8*)(lsBC + tl*32 + colh) = v;
    }
    __syncthreads();
    // prefetch next chunk's dt for this lane (speculative, dropped if OOB)
    {
      int snext = (s0 + SCHUNK < TT) ? (s0 + SCHUNK) : s0;
      int tp = rev ? (TT - 1 - snext) : snext;
      __builtin_prefetch(dt + ((size_t)(b*TT + tp))*DIN + d, 0, 0);
    }
#pragma unroll 1
    for (int sl = 0; sl < SCHUNK; ++sl) {
      int t = rev ? (TT - 1 - (s0 + sl)) : (s0 + sl);
      size_t mrow = (size_t)b*TT + t;
      float dtv = dt[mrow*DIN + d];
      float u   = bf2f(xi[mrow*DIN + d]);
      float du  = dtv * u;
      float y = 0.f;
#pragma unroll
      for (int n = 0; n < DSTATE; ++n) {
        float Bn = bf2f(lsBC[sl*32 + n]);
        float Cn = bf2f(lsBC[sl*32 + 16 + n]);
        float dA = __expf(dtv * Av[n]);
        h[n] = dA * h[n] + du * Bn;
        y += h[n] * Cn;
      }
      float z = xz[mrow*(2*DIN) + DIN + d];
      yz[mrow*DIN + d] = f2bf((y + u*Dd) * siluf(z));
    }
    __syncthreads();
  }
}

// ---------------------------------------------------------------------------
// LayerNorm over 512 channels, one wave per token row; affine; bf16 out.
// ---------------------------------------------------------------------------
__global__ __launch_bounds__(256) void lnK(
    const float* __restrict__ src, const float* __restrict__ lw,
    const float* __restrict__ lb, __bf16* __restrict__ dst)
{
  const int lane = threadIdx.x & 31, wid = threadIdx.x >> 5;
  const int row = blockIdx.x*8 + wid;
  const float* rp = src + (size_t)row * (2*CC);
  float v[16];
  float s = 0.f, q = 0.f;
#pragma unroll
  for (int k = 0; k < 16; ++k) {
    v[k] = rp[lane*16 + k];
    s += v[k]; q += v[k]*v[k];
  }
#pragma unroll
  for (int o = 16; o > 0; o >>= 1) {
    s += __shfl_xor(s, o, 32);
    q += __shfl_xor(q, o, 32);
  }
  const float mu = s * (1.f/512.f);
  const float var = q * (1.f/512.f) - mu*mu;
  const float rstd = rsqrtf(var + EPSF);
  __bf16* op = dst + (size_t)row * (2*CC);
#pragma unroll
  for (int k = 0; k < 16; ++k) {
    int c = lane*16 + k;
    op[c] = f2bf((v[k] - mu)*rstd*lw[c] + lb[c]);
  }
}

// ---------------------------------------------------------------------------
// Host orchestration
// ---------------------------------------------------------------------------
static void launch_gemm(const GemmP& p, hipStream_t s){
  dim3 g((p.M + 127)/128, (p.N + 63)/64);
  gemmK<<<g, 256, 0, s>>>(p);
}

extern "C" void kernel_launch(void* const* d_in, const int* in_sizes, int n_in,
                              void* d_out, int out_size, void* d_ws, size_t ws_size,
                              hipStream_t stream)
{
  (void)in_sizes; (void)n_in; (void)out_size; (void)ws_size;

  const float* x_in  = (const float*)d_in[0];
  // d_in[1] = style (unused by reference)
  const unsigned char* msk = (const unsigned char*)d_in[2];
  const float* cb_w  = (const float*)d_in[3];
  const float* cb_b  = (const float*)d_in[4];
  const float* gn_w  = (const float*)d_in[5];
  const float* gn_b  = (const float*)d_in[6];
  const float* ca_w  = (const float*)d_in[7];
  const float* gate_w= (const float*)d_in[8];
  const float* gate_b= (const float*)d_in[9];
  const float* ln_w  = (const float*)d_in[10];
  const float* ln_b  = (const float*)d_in[11];
  const float* proj_w= (const float*)d_in[12];
  struct Dir {
    const float *in_w, *conv_w, *conv_b, *xproj_w, *dt_w, *dt_b, *A_log, *D, *out_w;
  };
  Dir dirs[2];
  for (int i = 0; i < 2; ++i) {
    int o = 13 + i*9;
    dirs[i].in_w    = (const float*)d_in[o+0];
    dirs[i].conv_w  = (const float*)d_in[o+1];
    dirs[i].conv_b  = (const float*)d_in[o+2];
    dirs[i].xproj_w = (const float*)d_in[o+3];
    dirs[i].dt_w    = (const float*)d_in[o+4];
    dirs[i].dt_b    = (const float*)d_in[o+5];
    dirs[i].A_log   = (const float*)d_in[o+6];
    dirs[i].D       = (const float*)d_in[o+7];
    dirs[i].out_w   = (const float*)d_in[o+8];
  }

  // ---- workspace layout (~110 MB; resident in 192MB L2) ----
  char* w8 = (char*)d_ws;
  size_t off = 0;
  auto alloc = [&](size_t bytes) -> char* {
    char* p = w8 + off;
    off += (bytes + 255) & ~(size_t)255;
    return p;
  };
  float*  R0   = (float*) alloc((size_t)BB*CC*TT*4);   // 8MB conv ping
  float*  R1   = (float*) alloc((size_t)BB*CC*TT*4);   // 8MB conv pong
  float*  BI2  = R0;                                   // 16MB spans R0+R1 (reused)
  float*  STATS= (float*) alloc(256);
  __bf16* XT   = (__bf16*)alloc((size_t)MM*CC*2);
  float*  XZ   = (float*) alloc((size_t)MM*2*DIN*4);
  __bf16* XI   = (__bf16*)alloc((size_t)MM*DIN*2);
  __bf16* PROJ = (__bf16*)alloc((size_t)MM*48*2);
  float*  DT   = (float*) alloc((size_t)MM*DIN*4);
  __bf16* YZ   = (__bf16*)alloc((size_t)MM*DIN*2);
  __bf16* BIBF = (__bf16*)alloc((size_t)MM*2*CC*2);
  __bf16* LNBF = (__bf16*)alloc((size_t)MM*2*CC*2);
  __bf16* WIN[2]  = { (__bf16*)alloc(2*DIN*CC*2), (__bf16*)alloc(2*DIN*CC*2) };
  __bf16* WXP[2]  = { (__bf16*)alloc(48*DIN*2),   (__bf16*)alloc(48*DIN*2)   };
  __bf16* WDT[2]  = { (__bf16*)alloc(DIN*DTRANK*2),(__bf16*)alloc(DIN*DTRANK*2) };
  __bf16* WOUT[2] = { (__bf16*)alloc(CC*DIN*2),   (__bf16*)alloc(CC*DIN*2)   };
  __bf16* WGATE   = (__bf16*)alloc(2*CC*CC*2);
  __bf16* WCA     = (__bf16*)alloc(2*CC*2*CC*2);
  __bf16* WPROJ   = (__bf16*)alloc(CC*2*CC*2);

  auto cast = [&](const float* s, __bf16* d, int n){
    castK<<<(n + 255)/256, 256, 0, stream>>>(s, d, n);
  };
  for (int i = 0; i < 2; ++i) {
    cast(dirs[i].in_w,    WIN[i],  2*DIN*CC);
    cast(dirs[i].xproj_w, WXP[i],  48*DIN);
    cast(dirs[i].dt_w,    WDT[i],  DIN*DTRANK);
    cast(dirs[i].out_w,   WOUT[i], CC*DIN);
  }
  cast(gate_w, WGATE, 2*CC*CC);
  cast(ca_w,   WCA,   2*CC*2*CC);
  cast(proj_w, WPROJ, CC*2*CC);

  // ---- conv + GroupNorm + LeakyReLU + mask stack (DEPTH=4) ----
  const float* src = x_in;
  float* bufs[2] = { R0, R1 };
  for (int i = 0; i < 4; ++i) {
    hipMemsetAsync(STATS, 0, 256, stream);
    convgnK<<<dim3(TT/256, CC, BB), 256, 0, stream>>>(
        src, bufs[i & 1], cb_w + (size_t)i*CC*12, cb_b + i*CC, STATS);
    gnormK<<<(BB*CC*TT)/256, 256, 0, stream>>>(
        bufs[i & 1], STATS, gn_w + i*CC, gn_b + i*CC, msk);
    src = bufs[i & 1];
  }
  // final conv-stack output is bufs[1]
  transposeK<<<dim3(TT/32, CC/32, BB), dim3(32, 8), 0, stream>>>(bufs[1], XT);

  // ---- bidirectional Mamba (token order unflipped; direction handled by
  //      anti-causal conv + reverse scan; pointwise GEMMs are order-agnostic) ----
  for (int dir = 0; dir < 2; ++dir) {
    GemmP p{};
    // xz = xt @ in_w^T  (M=8192, N=1024, K=256) -> f32
    p = GemmP{}; p.A = XT; p.lda = CC; p.W = WIN[dir]; p.ldw = CC;
    p.outF = XZ; p.ldo = 2*DIN; p.M = MM; p.N = 2*DIN; p.K = CC;
    launch_gemm(p, stream);
    // depthwise conv + silu -> xi (bf16)
    mconvK<<<(MM*DIN)/256, 256, 0, stream>>>(XZ, dirs[dir].conv_w,
                                             dirs[dir].conv_b, XI, dir);
    // proj = xi @ xproj^T  (N=48) -> bf16
    p = GemmP{}; p.A = XI; p.lda = DIN; p.W = WXP[dir]; p.ldw = DIN;
    p.outB = PROJ; p.ldo = 48; p.M = MM; p.N = 48; p.K = DIN;
    launch_gemm(p, stream);
    // dt = softplus(proj[:, :16] @ dt_w^T + dt_b)  (K=16 zero-padded) -> f32
    p = GemmP{}; p.A = PROJ; p.lda = 48; p.W = WDT[dir]; p.ldw = DTRANK;
    p.outF = DT; p.ldo = DIN; p.bias = dirs[dir].dt_b; p.act = 1;
    p.M = MM; p.N = DIN; p.K = DTRANK;
    launch_gemm(p, stream);
    // selective scan (fwd / reverse) fused with y=(ys+u*D)*silu(z) -> bf16
    scanK<<<BB*(DIN/256), 256, 0, stream>>>(PROJ, DT, XI, XZ,
        dirs[dir].A_log, dirs[dir].D, YZ, dir);
    // out = yz @ out_w^T -> bf16 straight into bi[:, dir*256 : dir*256+256]
    p = GemmP{}; p.A = YZ; p.lda = DIN; p.W = WOUT[dir]; p.ldw = DIN;
    p.outB = BIBF + dir*CC; p.ldo = 2*CC; p.M = MM; p.N = CC; p.K = DIN;
    launch_gemm(p, stream);
  }

  // ---- gated fusion: gated = silu(grouped 1x1 conv(bi) + b) * bi -> BI2 ----
  for (int g = 0; g < 2; ++g) {
    GemmP p{};
    p.A = BIBF + g*CC; p.lda = 2*CC;
    p.W = WGATE + (size_t)g*CC*CC; p.ldw = CC;
    p.outF = BI2 + g*CC; p.ldo = 2*CC;
    p.bias = gate_b + g*CC; p.act = 2;
    p.mul = BIBF + g*CC; p.ldmul = 2*CC;
    p.M = MM; p.N = CC; p.K = CC;
    launch_gemm(p, stream);
  }
  // att = bi @ ca_w^T, accumulated onto gated -> BI2
  {
    GemmP p{};
    p.A = BIBF; p.lda = 2*CC; p.W = WCA; p.ldw = 2*CC;
    p.outF = BI2; p.ldo = 2*CC; p.accAdd = 1;
    p.M = MM; p.N = 2*CC; p.K = 2*CC;
    launch_gemm(p, stream);
  }
  // LayerNorm(512) -> bf16
  lnK<<<MM/8, 256, 0, stream>>>(BI2, ln_w, ln_b, LNBF);
  // fused = ln @ proj_w^T, transposed (B,C,T) masked store -> d_out (f32)
  {
    GemmP p{};
    p.A = LNBF; p.lda = 2*CC; p.W = WPROJ; p.ldw = 2*CC;
    p.outT = (float*)d_out; p.mask = msk; p.Tdim = TT;
    p.M = MM; p.N = CC; p.K = 2*CC;
    launch_gemm(p, stream);
  }
}
